// AtomicLinearTest_16844861735164
// MI455X (gfx1250) — matrix operand order repaired and checked by reading, not compiled, not run on hardware
//
#include <hip/hip_runtime.h>

#define IN_F   1024
#define OUT_F  1024
#define BATCHN 16384

// fused fallback tile config
#define BM 128
#define BN 128
#define BK 32
#define LDST 40   // padded LDS row stride in bf16 elements (80B = 20 banks -> conflict-free)

// async main-kernel tile config (64x64 per wave, 8 waves = 256x128 per block)
#define GBM 256
#define GBN 128

typedef __attribute__((ext_vector_type(16))) __bf16 v16bf;
typedef __attribute__((ext_vector_type(8)))  float  v8f;

union FragU { uint4 q[2]; v16bf v; };

static __device__ inline unsigned short f2bf_rne(float f) {
  unsigned u = __builtin_bit_cast(unsigned, f);
  u += 0x7FFFu + ((u >> 16) & 1u);           // round-to-nearest-even
  return (unsigned short)(u >> 16);
}
static __device__ inline float bf2f(unsigned short h) {
  unsigned u = ((unsigned)h) << 16;
  return __builtin_bit_cast(float, u);
}

// ---------------------------------------------------------------------------
// Prep: split fp32 -> bf16 hi/lo (hi = RNE(bf16), lo = RNE(bf16(f - hi)))
// ---------------------------------------------------------------------------
__global__ __launch_bounds__(256)
void split_bf16(const float* __restrict__ src,
                unsigned short* __restrict__ hi,
                unsigned short* __restrict__ lo, int n4) {
  int i = blockIdx.x * 256 + threadIdx.x;
  if (i >= n4) return;
  float4 f = ((const float4*)src)[i];
  unsigned short h0 = f2bf_rne(f.x), h1 = f2bf_rne(f.y),
                 h2 = f2bf_rne(f.z), h3 = f2bf_rne(f.w);
  uint2 ph; ph.x = h0 | ((unsigned)h1 << 16); ph.y = h2 | ((unsigned)h3 << 16);
  ((uint2*)hi)[i] = ph;
  unsigned short g0 = f2bf_rne(f.x - bf2f(h0)), g1 = f2bf_rne(f.y - bf2f(h1)),
                 g2 = f2bf_rne(f.z - bf2f(h2)), g3 = f2bf_rne(f.w - bf2f(h3));
  uint2 pl; pl.x = g0 | ((unsigned)g1 << 16); pl.y = g2 | ((unsigned)g3 << 16);
  ((uint2*)lo)[i] = pl;
}

// ---------------------------------------------------------------------------
// Async tile mover: ROWSx32 bf16 tile (row stride 1024 elems in global) ->
// padded LDS tile via global_load_async_to_lds_b128.
// CPT = 16B chunks per thread = ROWS*4/256.
// ---------------------------------------------------------------------------
template <int CPT>
static __device__ inline void async_tile(const unsigned short* __restrict__ base,
                                         int rowBase, int k0,
                                         unsigned short* ldsTile, int tid) {
#pragma unroll
  for (int i = 0; i < CPT; ++i) {
    int c    = i * 256 + tid;          // 16-byte chunk index
    int row  = c >> 2;
    int colE = (c & 3) * 8;            // element column: 0,8,16,24
    unsigned voff = (unsigned)(((rowBase + row) * IN_F + k0 + colE) * 2);
    unsigned loff = (unsigned)(size_t)(ldsTile + row * LDST + colE);
    asm volatile("global_load_async_to_lds_b128 %0, %1, %2"
                 :: "v"(loff), "v"(voff), "s"(base) : "memory");
  }
}

// ---------------------------------------------------------------------------
// Main GEMM: y = x @ W^T + b, 3-product bf16 WMMA, 64x64 per wave,
// async double-buffered LDS (120 KB)
// ---------------------------------------------------------------------------
__global__ __launch_bounds__(256)
void gemm_bf16x3_async(const unsigned short* __restrict__ xh,
                       const unsigned short* __restrict__ xl,
                       const unsigned short* __restrict__ wh,
                       const unsigned short* __restrict__ wl,
                       const float* __restrict__ bias,
                       float* __restrict__ y) {
  __shared__ __align__(16) unsigned short ldsX[2][2][GBM * LDST]; // [buf][hi/lo]
  __shared__ __align__(16) unsigned short ldsW[2][2][GBN * LDST]; // [buf][hi/lo]

  const int tid   = threadIdx.x;
  const int lane  = tid & 31;
  const int wave  = tid >> 5;
  const int waveM = wave & 3;       // 4 waves along M -> 64-row slabs
  const int waveN = wave >> 2;      // 2 waves along N -> 64-col slabs

  const int mBlock = blockIdx.y * GBM;
  const int nBlock = blockIdx.x * GBN;

  const int half = lane >> 4;
  const int l16  = lane & 15;
  const int aCol = half ? 8 : 0;    // A frag chunks at aCol, aCol+16
  const int bCol = half ? 8 : 0;    // B frag chunks at bCol, bCol+16

  v8f acc[4][4] = {};

  // prologue: stage chunk 0 into buffer 0 (12 async instrs / wave)
  async_tile<4>(xh, mBlock, 0, &ldsX[0][0][0], tid);
  async_tile<4>(xl, mBlock, 0, &ldsX[0][1][0], tid);
  async_tile<2>(wh, nBlock, 0, &ldsW[0][0][0], tid);
  async_tile<2>(wl, nBlock, 0, &ldsW[0][1][0], tid);

  for (int k0 = 0, kIdx = 0; k0 < IN_F; k0 += BK, ++kIdx) {
    const int cur = kIdx & 1;
    __syncthreads();                       // buffer cur^1 free for refill
    if (k0 + BK < IN_F) {                  // uniform branch: EXEC stays all-ones
      async_tile<4>(xh, mBlock, k0 + BK, &ldsX[cur ^ 1][0][0], tid);
      async_tile<4>(xl, mBlock, k0 + BK, &ldsX[cur ^ 1][1][0], tid);
      async_tile<2>(wh, nBlock, k0 + BK, &ldsW[cur ^ 1][0][0], tid);
      async_tile<2>(wl, nBlock, k0 + BK, &ldsW[cur ^ 1][1][0], tid);
      // 12 newly issued outstanding; <=12 left => chunk k's 12 completed (in-order)
      asm volatile("s_wait_asynccnt 0xc" ::: "memory");
    } else {
      asm volatile("s_wait_asynccnt 0x0" ::: "memory");
    }
    __syncthreads();                       // all waves' chunk-k data in LDS

    FragU aH[4], aL[4];
#pragma unroll
    for (int m = 0; m < 4; ++m) {
      const unsigned short* pH = &ldsX[cur][0][(waveM * 64 + m * 16 + l16) * LDST];
      const unsigned short* pL = &ldsX[cur][1][(waveM * 64 + m * 16 + l16) * LDST];
      aH[m].q[0] = *(const uint4*)(pH + aCol);
      aH[m].q[1] = *(const uint4*)(pH + aCol + 16);
      aL[m].q[0] = *(const uint4*)(pL + aCol);
      aL[m].q[1] = *(const uint4*)(pL + aCol + 16);
    }
#pragma unroll
    for (int n = 0; n < 4; ++n) {
      FragU bH, bL;
      const unsigned short* pH = &ldsW[cur][0][(waveN * 64 + n * 16 + l16) * LDST];
      const unsigned short* pL = &ldsW[cur][1][(waveN * 64 + n * 16 + l16) * LDST];
      bH.q[0] = *(const uint4*)(pH + bCol);
      bH.q[1] = *(const uint4*)(pH + bCol + 16);
      bL.q[0] = *(const uint4*)(pL + bCol);
      bL.q[1] = *(const uint4*)(pL + bCol + 16);
#pragma unroll
      for (int m = 0; m < 4; ++m) {
        acc[m][n] = __builtin_amdgcn_wmma_f32_16x16x32_bf16(
            false, aH[m].v, false, bH.v, (short)0, acc[m][n], false, false);
        acc[m][n] = __builtin_amdgcn_wmma_f32_16x16x32_bf16(
            false, aH[m].v, false, bL.v, (short)0, acc[m][n], false, false);
        acc[m][n] = __builtin_amdgcn_wmma_f32_16x16x32_bf16(
            false, aL[m].v, false, bH.v, (short)0, acc[m][n], false, false);
      }
    }
  }

  // epilogue: bias + store (C/D layout: VGPR r -> M=r lanes0-15, M=8+r lanes16-31)
#pragma unroll
  for (int n = 0; n < 4; ++n) {
    int ocol = nBlock + waveN * 64 + n * 16 + l16;
    float bv = bias[ocol];
#pragma unroll
    for (int m = 0; m < 4; ++m) {
      int rbase = mBlock + waveM * 64 + m * 16 + (half ? 8 : 0);
#pragma unroll
      for (int r = 0; r < 8; ++r) {
        y[(size_t)(rbase + r) * OUT_F + ocol] = acc[m][n][r] + bv;
      }
    }
  }
}

// ---------------------------------------------------------------------------
// Fallback (fused) kernel if workspace is too small for the split buffers
// ---------------------------------------------------------------------------
__global__ __launch_bounds__(256)
void linear_bf16x3_fused(const float* __restrict__ x,
                         const float* __restrict__ w,
                         const float* __restrict__ bias,
                         float* __restrict__ y) {
  __shared__ __align__(16) unsigned short xh[BM * LDST];
  __shared__ __align__(16) unsigned short xl[BM * LDST];
  __shared__ __align__(16) unsigned short wh[BN * LDST];
  __shared__ __align__(16) unsigned short wl[BN * LDST];

  const int tid   = threadIdx.x;
  const int lane  = tid & 31;
  const int wave  = tid >> 5;
  const int waveM = wave & 3;
  const int waveN = wave >> 2;
  const int mBlock = blockIdx.y * BM;
  const int nBlock = blockIdx.x * BN;
  const int half = lane >> 4;
  const int l16  = lane & 15;
  const int aCol = half ? 8 : 0;
  const int bCol = half ? 8 : 0;

  v8f acc[2][4] = {};

  for (int k0 = 0; k0 < IN_F; k0 += BK) {
    float4 xv[4], wv[4];
#pragma unroll
    for (int i = 0; i < 4; ++i) {
      int flat = (i * 256 + tid) * 4;
      int r = flat >> 5;
      int c = flat & 31;
      xv[i] = *(const float4*)(x + (size_t)(mBlock + r) * IN_F + k0 + c);
      wv[i] = *(const float4*)(w + (size_t)(nBlock + r) * IN_F + k0 + c);
    }
    __syncthreads();
#pragma unroll
    for (int i = 0; i < 4; ++i) {
      int flat = (i * 256 + tid) * 4;
      int r = flat >> 5;
      int c = flat & 31;
      {
        float4 f = xv[i];
        unsigned short h0 = f2bf_rne(f.x), h1 = f2bf_rne(f.y),
                       h2 = f2bf_rne(f.z), h3 = f2bf_rne(f.w);
        uint2 ph; ph.x = h0 | ((unsigned)h1 << 16); ph.y = h2 | ((unsigned)h3 << 16);
        *(uint2*)(&xh[r * LDST + c]) = ph;
        unsigned short g0 = f2bf_rne(f.x - bf2f(h0)), g1 = f2bf_rne(f.y - bf2f(h1)),
                       g2 = f2bf_rne(f.z - bf2f(h2)), g3 = f2bf_rne(f.w - bf2f(h3));
        uint2 pl; pl.x = g0 | ((unsigned)g1 << 16); pl.y = g2 | ((unsigned)g3 << 16);
        *(uint2*)(&xl[r * LDST + c]) = pl;
      }
      {
        float4 f = wv[i];
        unsigned short h0 = f2bf_rne(f.x), h1 = f2bf_rne(f.y),
                       h2 = f2bf_rne(f.z), h3 = f2bf_rne(f.w);
        uint2 ph; ph.x = h0 | ((unsigned)h1 << 16); ph.y = h2 | ((unsigned)h3 << 16);
        *(uint2*)(&wh[r * LDST + c]) = ph;
        unsigned short g0 = f2bf_rne(f.x - bf2f(h0)), g1 = f2bf_rne(f.y - bf2f(h1)),
                       g2 = f2bf_rne(f.z - bf2f(h2)), g3 = f2bf_rne(f.w - bf2f(h3));
        uint2 pl; pl.x = g0 | ((unsigned)g1 << 16); pl.y = g2 | ((unsigned)g3 << 16);
        *(uint2*)(&wl[r * LDST + c]) = pl;
      }
    }
    __syncthreads();

    FragU aH[2], aL[2];
#pragma unroll
    for (int m = 0; m < 2; ++m) {
      const unsigned short* pH = &xh[(waveM * 32 + m * 16 + l16) * LDST];
      const unsigned short* pL = &xl[(waveM * 32 + m * 16 + l16) * LDST];
      aH[m].q[0] = *(const uint4*)(pH + aCol);
      aH[m].q[1] = *(const uint4*)(pH + aCol + 16);
      aL[m].q[0] = *(const uint4*)(pL + aCol);
      aL[m].q[1] = *(const uint4*)(pL + aCol + 16);
    }
#pragma unroll
    for (int n = 0; n < 4; ++n) {
      FragU bH, bL;
      const unsigned short* pH = &wh[(waveN * 64 + n * 16 + l16) * LDST];
      const unsigned short* pL = &wl[(waveN * 64 + n * 16 + l16) * LDST];
      bH.q[0] = *(const uint4*)(pH + bCol);
      bH.q[1] = *(const uint4*)(pH + bCol + 16);
      bL.q[0] = *(const uint4*)(pL + bCol);
      bL.q[1] = *(const uint4*)(pL + bCol + 16);
#pragma unroll
      for (int m = 0; m < 2; ++m) {
        acc[m][n] = __builtin_amdgcn_wmma_f32_16x16x32_bf16(
            false, aH[m].v, false, bH.v, (short)0, acc[m][n], false, false);
        acc[m][n] = __builtin_amdgcn_wmma_f32_16x16x32_bf16(
            false, aH[m].v, false, bL.v, (short)0, acc[m][n], false, false);
        acc[m][n] = __builtin_amdgcn_wmma_f32_16x16x32_bf16(
            false, aL[m].v, false, bH.v, (short)0, acc[m][n], false, false);
      }
    }
  }

#pragma unroll
  for (int n = 0; n < 4; ++n) {
    int ocol = nBlock + waveN * 64 + n * 16 + l16;
    float bv = bias[ocol];
#pragma unroll
    for (int m = 0; m < 2; ++m) {
      int rbase = mBlock + waveM * 32 + m * 16 + (half ? 8 : 0);
#pragma unroll
      for (int r = 0; r < 8; ++r) {
        y[(size_t)(rbase + r) * OUT_F + ocol] = acc[m][n][r] + bv;
      }
    }
  }
}

extern "C" void kernel_launch(void* const* d_in, const int* in_sizes, int n_in,
                              void* d_out, int out_size, void* d_ws, size_t ws_size,
                              hipStream_t stream) {
  const float* x  = (const float*)d_in[0];
  const float* w  = (const float*)d_in[1];
  const float* b  = (const float*)d_in[2];
  float*       y  = (float*)d_out;

  const size_t xElems = (size_t)BATCHN * IN_F;   // 16 Mi
  const size_t wElems = (size_t)OUT_F * IN_F;    // 1 Mi
  const size_t need   = (xElems + wElems) * 2 * sizeof(unsigned short); // ~68 MB

  dim3 block(256);

  if (ws_size >= need) {
    unsigned short* xhg = (unsigned short*)d_ws;
    unsigned short* xlg = xhg + xElems;
    unsigned short* whg = xlg + xElems;
    unsigned short* wlg = whg + wElems;

    int xn4 = (int)(xElems / 4);
    int wn4 = (int)(wElems / 4);
    split_bf16<<<dim3((xn4 + 255) / 256), block, 0, stream>>>(x, xhg, xlg, xn4);
    split_bf16<<<dim3((wn4 + 255) / 256), block, 0, stream>>>(w, whg, wlg, wn4);

    dim3 grid(OUT_F / GBN, BATCHN / GBM);   // (8, 64)
    gemm_bf16x3_async<<<grid, block, 0, stream>>>(xhg, xlg, whg, wlg, b, y);
  } else {
    dim3 grid(OUT_F / BN, BATCHN / BM);     // (8, 128)
    linear_bf16x3_fused<<<grid, block, 0, stream>>>(x, w, b, y);
  }
}
